// LogNeuralBackflow_72370198937642
// MI455X (gfx1250) — matrix operand order repaired and checked by reading, not compiled, not run on hardware
//
#include <hip/hip_runtime.h>
#include <hip/hip_bf16.h>
#include <math.h>

// Problem constants (match reference)
#define B_TOT 16384
#define NORB  64
#define NF    32
#define HID   512
#define OUT2  (NORB * NF)   // 2048

typedef __attribute__((ext_vector_type(16))) _Float16 v16h;
typedef __attribute__((ext_vector_type(8)))  _Float16 v8h;
typedef __attribute__((ext_vector_type(8)))  float    v8f;
typedef __attribute__((ext_vector_type(4)))  int      v4i;

// Workspace layout (bytes)
#define W1H_OFF  0u
#define W2H_OFF  (64u * 512u * 2u)                       // 65536
#define HWS_OFF  (W2H_OFF + 512u * 2048u * 2u)           // 2162688
// total ws used: ~18.9 MB

// Branch-free fast tanh: 1 - 2/(e^{2x}+1).  Saturates correctly at +-1,
// ~1 ulp rcp error is far below the f16 quantization applied right after.
__device__ __forceinline__ float fast_tanh(float x) {
    const float e = __expf(2.0f * x);
    return 1.0f - 2.0f * __builtin_amdgcn_rcpf(e + 1.0f);
}

// ---------------------------------------------------------------------------
// Kernel 1: f32 -> f16 weight conversion (one-shot, bandwidth trivial)
// ---------------------------------------------------------------------------
__global__ void __launch_bounds__(256)
convert_weights(const float* __restrict__ W1, const float* __restrict__ W2,
                _Float16* __restrict__ W1h, _Float16* __restrict__ W2h) {
    const int t1 = 64 * 512;
    const int t2 = 512 * 2048;
    for (int i = blockIdx.x * blockDim.x + threadIdx.x; i < t1 + t2;
         i += gridDim.x * blockDim.x) {
        if (i < t1) W1h[i] = (_Float16)W1[i];
        else        W2h[i - t1] = (_Float16)W2[i - t1];
    }
}

// ---------------------------------------------------------------------------
// Kernel 2: h = tanh(n @ W1 + b1), stored f16.
// One wave = one 16(batch) x 16(hid) f32 tile, K=64 as two 16x16x32 WMMAs.
// A-frag layout (ISA 7.12.2, 16-bit A 16x32): lanes 0-15 row=lane, K={0..7,16..23};
// lanes 16-31 row=lane-16, K={8..15,24..31}.  The two 8-wide K chunks are
// contiguous ints in n -> four b128 loads per k-step.
// B-frag: lane holds N col = lane&15, the same 16 K values as its A-frag (stride HID).
// ---------------------------------------------------------------------------
__global__ void __launch_bounds__(256)
hidden_gemm(const int* __restrict__ n, const _Float16* __restrict__ W1h,
            const float* __restrict__ b1, _Float16* __restrict__ hws) {
    const int tid  = threadIdx.x;
    const int lane = tid & 31;
    const int w    = blockIdx.x * 8 + (tid >> 5);   // global wave id
    const int tile_m = w >> 5;                      // 0..1023 (B/16)
    const int tile_n = w & 31;                      // 0..31   (HID/16)
    const int row0 = tile_m * 16;
    const int n0   = tile_n * 16;

    const int mrow = lane & 15;
    const int s    = (lane < 16) ? 0 : 8;
    const int* nrow = n + (size_t)(row0 + mrow) * NORB;

    v8f acc = {};
#pragma unroll
    for (int ks = 0; ks < 2; ++ks) {
        const int kb = ks * 32;
        const v4i q0 = *(const v4i*)(nrow + kb + s);          // K = base+0..3
        const v4i q1 = *(const v4i*)(nrow + kb + s + 4);      // K = base+4..7
        const v4i q2 = *(const v4i*)(nrow + kb + 16 + s);     // K = base+16..19
        const v4i q3 = *(const v4i*)(nrow + kb + 16 + s + 4); // K = base+20..23
        v16h a;
#pragma unroll
        for (int i = 0; i < 4; ++i) {
            a[i]      = (_Float16)(float)q0[i];   // n in {0,1}: exact
            a[i + 4]  = (_Float16)(float)q1[i];
            a[i + 8]  = (_Float16)(float)q2[i];
            a[i + 12] = (_Float16)(float)q3[i];
        }
        v16h b;
#pragma unroll
        for (int i = 0; i < 16; ++i) {
            const int k = kb + s + (i < 8 ? i : 8 + i);
            b[i] = W1h[(size_t)k * HID + n0 + (lane & 15)];
        }
        acc = __builtin_amdgcn_wmma_f32_16x16x32_f16(
                  false, a, false, b, (short)0, acc, false, false);
    }

    const int col = n0 + (lane & 15);
    const float bias = b1[col];
#pragma unroll
    for (int v = 0; v < 8; ++v) {
        const int m = (lane < 16) ? v : (v + 8);       // C/D layout (ISA 7.12.2)
        const float hv = fast_tanh(acc[v] + bias);
        hws[(size_t)(row0 + m) * HID + col] = (_Float16)hv;
    }
}

// ---------------------------------------------------------------------------
// Kernel 3: fused  F = h @ W2 + b2  ->  gather occupied rows of (M + F)
//           -> batched 32x32 LU slogdet.  One workgroup = 16 batch rows.
// LDS: A matrices 16 x (32x33 padded) f32 = 67584 B
//      h rows     16 x 512 f16          = 16384 B
//      rank table 16 x 64 i8            =  1024 B   (total 84992 B < 320 KB/WGP)
// h staging uses the gfx1250 async global->LDS path (ASYNCcnt).
// ---------------------------------------------------------------------------
__global__ void __launch_bounds__(256)
backflow_slogdet(const int* __restrict__ n, const float* __restrict__ Mflat,
                 const float* __restrict__ b2, const _Float16* __restrict__ W2h,
                 const _Float16* __restrict__ hws, float* __restrict__ out) {
    extern __shared__ char smem[];
    float*       A_lds    = (float*)smem;                          // 16*1056
    _Float16*    h_lds    = (_Float16*)(smem + 16 * 1056 * 4);     // 16*512
    signed char* rank_tbl = (signed char*)(smem + 16 * 1056 * 4 + 16 * 512 * 2);

    const int tid  = threadIdx.x;
    const int lane = tid & 31;
    const int wid  = tid >> 5;      // 0..7
    const int b0   = blockIdx.x * 16;

    // Stage h rows (16 KB) with async global->LDS b128 copies (4 per thread).
    // GVS mode: LDS address from the vdst VGPR; global address = SGPR base + VGPR byte offset.
    {
        const _Float16* gsrc = hws + (size_t)b0 * HID;
        const unsigned lds_base = (unsigned)(uintptr_t)h_lds;  // low 32b = LDS offset
#pragma unroll
        for (int i = 0; i < 4; ++i) {
            const unsigned off = (unsigned)(i * 4096 + tid * 16);
            asm volatile("global_load_async_to_lds_b128 %0, %1, %2"
                         :
                         : "v"(lds_base + off), "v"(off), "s"(gsrc)
                         : "memory");
        }
        asm volatile("s_wait_asynccnt 0" ::: "memory");
    }
    // Occupancy rank table: rank_tbl[m][o] = #occupied orbitals < o (or -1)
    if (tid < 16) {
        const int* nrow = n + (size_t)(b0 + tid) * NORB;
        int r = 0;
        for (int o = 0; o < NORB; ++o)
            rank_tbl[tid * NORB + o] = nrow[o] ? (signed char)(r++) : (signed char)(-1);
    }
    __syncthreads();

    // Prebuild all 16 A-fragments of h (K=512 in 32-chunks), kept in VGPRs.
    const int mrow = lane & 15;
    const int s    = (lane < 16) ? 0 : 8;
    v16h af[16];
#pragma unroll
    for (int ks = 0; ks < 16; ++ks) {
        const int kb = ks * 32;
        v8h lo = *(const v8h*)(h_lds + mrow * HID + kb + s);        // ds_load_b128
        v8h hi = *(const v8h*)(h_lds + mrow * HID + kb + 16 + s);   // ds_load_b128
#pragma unroll
        for (int i = 0; i < 8; ++i) { af[ks][i] = lo[i]; af[ks][i + 8] = hi[i]; }
    }

    // Each wave sweeps 16 of the 128 column tiles of W2 (N=2048).
    for (int t = 0; t < 16; ++t) {
        const int c0 = (wid * 16 + t) * 16;
        const int c  = c0 + (lane & 15);
        // Warm WGP cache for the next column tile (global_prefetch_b8).
        if (t < 15)
            __builtin_prefetch(W2h + (size_t)lane * OUT2 + c0 + 16, 0, 3);
        const float addc = b2[c] + Mflat[c];   // col c == orbital*32 + f == flat M index
        v8f acc = {};
#pragma unroll
        for (int ks = 0; ks < 16; ++ks) {
            v16h bm;
#pragma unroll
            for (int i = 0; i < 16; ++i) {
                const int k = ks * 32 + s + (i < 8 ? i : 8 + i);     // B col = lane&15, k matches A
                bm[i] = W2h[(size_t)k * OUT2 + c];
            }
            acc = __builtin_amdgcn_wmma_f32_16x16x32_f16(
                      false, af[ks], false, bm, (short)0, acc, false, false);
        }
        // Scatter occupied rows into the batched A matrices
        const int o = c >> 5, f = c & 31;
#pragma unroll
        for (int v = 0; v < 8; ++v) {
            const int m  = (lane < 16) ? v : (v + 8);
            const int rk = rank_tbl[m * NORB + o];
            if (rk >= 0) A_lds[m * 1056 + rk * 33 + f] = acc[v] + addc;
        }
    }
    __syncthreads();

    // Partial-pivot LU slogdet: one wave per 32x32 matrix, 2 matrices/wave.
    for (int mm = wid; mm < 16; mm += 8) {
        float* A = A_lds + mm * 1056;
        float sign = 1.f, logab = 0.f;
        for (int k = 0; k < 32; ++k) {
            // argmax |A[i][k]| over i >= k (butterfly reduce -> uniform result)
            float val = (lane >= k) ? fabsf(A[lane * 33 + k]) : -1.f;
            int   idx = lane;
#pragma unroll
            for (int off = 16; off > 0; off >>= 1) {
                const float ov = __shfl_xor(val, off, 32);
                const int   oi = __shfl_xor(idx, off, 32);
                if (ov > val) { val = ov; idx = oi; }
            }
            if (idx != k) {                       // uniform branch: swap rows k<->idx
                const float t0 = A[k * 33 + lane];
                const float t1 = A[idx * 33 + lane];
                A[k * 33 + lane]   = t1;
                A[idx * 33 + lane] = t0;
                sign = -sign;
            }
            __builtin_amdgcn_wave_barrier();
            const float piv  = A[k * 33 + k];
            const float rpiv = __builtin_amdgcn_rcpf(piv);  // 1 rcp instead of 31 divs
            logab += __logf(fabsf(piv));
            if (piv < 0.f) sign = -sign;
            if (lane > k) {                       // lane = row i, serial over j
                const float mfac = A[lane * 33 + k] * rpiv;
                for (int j = k + 1; j < 32; ++j)
                    A[lane * 33 + j] -= mfac * A[k * 33 + j];
            }
            __builtin_amdgcn_wave_barrier();
        }
        if (lane == 0) {
            const int b = b0 + mm;
            out[2 * b]     = logab;                                   // Re
            out[2 * b + 1] = (sign < 0.f) ? 3.14159265358979323846f   // Im = pi iff det<0
                                          : 0.f;
        }
    }
}

// ---------------------------------------------------------------------------
extern "C" void kernel_launch(void* const* d_in, const int* in_sizes, int n_in,
                              void* d_out, int out_size, void* d_ws, size_t ws_size,
                              hipStream_t stream) {
    (void)in_sizes; (void)n_in; (void)out_size; (void)ws_size;
    const int*   n_occ = (const int*)d_in[0];
    const float* M     = (const float*)d_in[1];
    const float* W1    = (const float*)d_in[2];
    const float* b1    = (const float*)d_in[3];
    const float* W2    = (const float*)d_in[4];
    const float* b2    = (const float*)d_in[5];
    float* out = (float*)d_out;

    char* ws = (char*)d_ws;
    _Float16* W1h = (_Float16*)(ws + W1H_OFF);
    _Float16* W2h = (_Float16*)(ws + W2H_OFF);
    _Float16* hws = (_Float16*)(ws + HWS_OFF);

    const size_t smem3 = 16 * 1056 * 4 + 16 * HID * 2 + 16 * NORB;  // 84992 B
    // allow >64KB dynamic LDS (CDNA5: up to 320KB per workgroup)
    hipFuncSetAttribute((const void*)backflow_slogdet,
                        hipFuncAttributeMaxDynamicSharedMemorySize, (int)smem3);

    convert_weights<<<1024, 256, 0, stream>>>(W1, W2, W1h, W2h);
    // (B/16)*(HID/16) = 32768 wave-tiles, 8 waves per block
    hidden_gemm<<<(B_TOT / 16) * (HID / 16) / 8, 256, 0, stream>>>(n_occ, W1h, b1, hws);
    backflow_slogdet<<<B_TOT / 16, 256, smem3, stream>>>(n_occ, M, b2, W2h, hws, out);
}
